// TripletHardLoss_67207648247978
// MI455X (gfx1250) — compile-verified, matrix operand order checked by reading, not run on hardware
//
#include <hip/hip_runtime.h>
#include <hip/hip_bf16.h>

typedef _Float16 v16h __attribute__((ext_vector_type(16)));
typedef float    v8f  __attribute__((ext_vector_type(8)));

static constexpr int   N_ROWS = 4096;
static constexpr int   N_DIM  = 2048;
static constexpr float MARGIN = 0.3f;

// ---------------------------------------------------------------------------
// Kernel 1: fp32 -> f16 copy of X, fp32 row norms, init per-row atomic cells.
// ---------------------------------------------------------------------------
__global__ __launch_bounds__(256) void prep_kernel(const float* __restrict__ X,
                                                   _Float16* __restrict__ Xh,
                                                   float* __restrict__ sq,
                                                   int* __restrict__ apB,
                                                   int* __restrict__ anB) {
    const int row = blockIdx.x;
    const float*  xr = X  + (size_t)row * N_DIM;
    _Float16*     hr = Xh + (size_t)row * N_DIM;
    float acc = 0.0f;
    for (int k = threadIdx.x; k < N_DIM; k += 256) {
        float v = xr[k];
        hr[k] = (_Float16)v;
        acc = fmaf(v, v, acc);
    }
    #pragma unroll
    for (int off = 16; off > 0; off >>= 1) acc += __shfl_xor(acc, off, 32);
    __shared__ float s[8];
    const int wave = threadIdx.x >> 5, lane = threadIdx.x & 31;
    if (lane == 0) s[wave] = acc;
    __syncthreads();
    if (threadIdx.x == 0) {
        float t = 0.0f;
        #pragma unroll
        for (int w = 0; w < 8; ++w) t += s[w];
        sq[row]  = t;
        apB[row] = 0;            // bits of 0.0f
        anB[row] = 0x7F800000;   // bits of +inf
    }
}

// ---------------------------------------------------------------------------
// Kernel 2: fused Gram tile (WMMA f16 -> f32) + distance + masked row max/min.
// Double-buffered LDS staged with GLOBAL_LOAD_ASYNC_TO_LDS_B128 (ASYNCcnt).
// Block tile 128x128, 8 waves (4 M x 2 N), wave tile 32x64 (2x4 fragments).
// ---------------------------------------------------------------------------
#define ASYNC_B128(ldsaddr, gaddr)                                             \
    asm volatile("global_load_async_to_lds_b128 %0, %1, off"                   \
                 :: "v"(ldsaddr), "v"(gaddr) : "memory")

__global__ __launch_bounds__(256) void gram_kernel(const _Float16* __restrict__ Xh,
                                                   const float* __restrict__ sq,
                                                   const int* __restrict__ Y,
                                                   int* __restrict__ apB,
                                                   int* __restrict__ anB) {
    // Row stride padded to 40 halves (80 B = 5x16 B): fragment loads from 16
    // consecutive rows hit 16 distinct bank groups, 16 B aligned.
    __shared__ __align__(16) _Float16 As[2][128 * 40];   // 2 x 10240 B
    __shared__ __align__(16) _Float16 Bs[2][128 * 40];

    const int tid  = threadIdx.x;
    const int lane = tid & 31;
    const int wave = tid >> 5;
    const int wm   = wave >> 1;          // 0..3 : M sub-tile
    const int wn   = wave & 1;           // 0..1 : N sub-tile
    const int l16  = lane & 15;
    const int g    = lane >> 4;          // K-group select (ISA A/B lane layout)
    const int rowBase = blockIdx.y * 128;
    const int colBase = blockIdx.x * 128;

    v8f acc[2][4];
    const v8f vzero = {};
    #pragma unroll
    for (int mb = 0; mb < 2; ++mb)
        #pragma unroll
        for (int jb = 0; jb < 4; ++jb) acc[mb][jb] = vzero;

    // ---- async staging setup: 4 x b128 per thread per k-step ----
    // thread -> (row r0, 16B segment s0) and (row r0+64, segment s0)
    const int r0 = tid >> 2;
    const int s0 = tid & 3;
    const _Float16* gA = Xh + (size_t)(rowBase + r0) * N_DIM + s0 * 8;
    const _Float16* gB = Xh + (size_t)(colBase + r0) * N_DIM + s0 * 8;
    // LDS aperture occupies flat addr bits [63:32]; low 32 bits are the
    // workgroup-relative LDS byte offset the async op consumes.
    const unsigned int lA0 = (unsigned int)(uintptr_t)&As[0][r0 * 40 + s0 * 8];
    const unsigned int lA1 = (unsigned int)(uintptr_t)&As[1][r0 * 40 + s0 * 8];
    const unsigned int lB0 = (unsigned int)(uintptr_t)&Bs[0][r0 * 40 + s0 * 8];
    const unsigned int lB1 = (unsigned int)(uintptr_t)&Bs[1][r0 * 40 + s0 * 8];
    constexpr int ROW64   = 64 * N_DIM;   // +64 rows in Xh (halves)
    constexpr unsigned SEG64 = 64 * 80;   // +64 rows in a padded LDS tile (bytes)

    // prologue: tile 0 -> buffer 0
    ASYNC_B128(lA0,         gA);
    ASYNC_B128(lA0 + SEG64, gA + ROW64);
    ASYNC_B128(lB0,         gB);
    ASYNC_B128(lB0 + SEG64, gB + ROW64);

    union FragH { uint4 q[2]; v16h h; };
    int p = 0;
    for (int kk = 0; kk < N_DIM; kk += 32) {
        if (kk + 32 < N_DIM) {
            // issue next tile into the other buffer (its readers finished
            // before the barrier that ended the previous iteration)
            const unsigned int nA = (p ? lA0 : lA1);
            const unsigned int nB = (p ? lB0 : lB1);
            const int ko = kk + 32;
            ASYNC_B128(nA,         gA + ko);
            ASYNC_B128(nA + SEG64, gA + ko + ROW64);
            ASYNC_B128(nB,         gB + ko);
            ASYNC_B128(nB + SEG64, gB + ko + ROW64);
            // my 4 oldest (current tile) done; 4 newest still in flight
            asm volatile("s_wait_asynccnt 0x4" ::: "memory");
        } else {
            asm volatile("s_wait_asynccnt 0x0" ::: "memory");
        }
        __syncthreads();   // all waves' async copies for buffer p visible

        // Fragment loads: lane -> (row = l16, K = 8*g + {0..7} U {16+8*g..}).
        FragH af[2], bf[4];
        #pragma unroll
        for (int mb = 0; mb < 2; ++mb) {
            const _Float16* q = &As[p][(wm * 32 + mb * 16 + l16) * 40 + 8 * g];
            af[mb].q[0] = *(const uint4*)q;
            af[mb].q[1] = *(const uint4*)(q + 16);
        }
        #pragma unroll
        for (int jb = 0; jb < 4; ++jb) {
            const _Float16* q = &Bs[p][(wn * 64 + jb * 16 + l16) * 40 + 8 * g];
            bf[jb].q[0] = *(const uint4*)q;
            bf[jb].q[1] = *(const uint4*)(q + 16);
        }

        #pragma unroll
        for (int mb = 0; mb < 2; ++mb)
            #pragma unroll
            for (int jb = 0; jb < 4; ++jb)
                acc[mb][jb] = __builtin_amdgcn_wmma_f32_16x16x32_f16(
                    /*neg_a=*/false, af[mb].h, /*neg_b=*/false, bf[jb].h,
                    /*c_mod=*/(short)0, acc[mb][jb],
                    /*reuse_a=*/false, /*reuse_b=*/false);

        __syncthreads();   // all reads of buffer p done before it is refilled
        p ^= 1;
    }

    // -------- fused epilogue: distance, label mask, row max/min --------
    const float INF = __int_as_float(0x7F800000);
    float sqj[4]; int yj[4];
    #pragma unroll
    for (int jb = 0; jb < 4; ++jb) {
        const int j = colBase + wn * 64 + jb * 16 + l16;  // C/D: N = lane&15
        sqj[jb] = sq[j];
        yj[jb]  = Y[j];
    }
    #pragma unroll
    for (int mb = 0; mb < 2; ++mb) {
        #pragma unroll
        for (int r = 0; r < 8; ++r) {
            const int i   = rowBase + wm * 32 + mb * 16 + r + 8 * g;  // M = r + 8*(lane>=16)
            const float sqi = sq[i];
            const int   yi  = Y[i];
            float apv = -1.0f;   // distances >= 0; diagonal guarantees ap >= 0
            float anv = INF;
            #pragma unroll
            for (int jb = 0; jb < 4; ++jb) {
                const float gv = acc[mb][jb][r];
                float d2 = sqi + sqj[jb] - 2.0f * gv;
                d2 = fmaxf(d2, 0.0f);
                const float d = sqrtf(d2);
                const bool same = (yi == yj[jb]);
                apv = fmaxf(apv, same ? d : -1.0f);
                anv = fminf(anv, same ? INF : d);
            }
            #pragma unroll
            for (int off = 8; off > 0; off >>= 1) {
                apv = fmaxf(apv, __shfl_xor(apv, off, 16));
                anv = fminf(anv, __shfl_xor(anv, off, 16));
            }
            if (l16 == 0) {
                // non-negative floats (+inf included) order like their int bits
                atomicMax(&apB[i], __float_as_int(apv));
                atomicMin(&anB[i], __float_as_int(anv));
            }
        }
    }
}

// ---------------------------------------------------------------------------
// Kernel 3: mean hinge over rows.
// ---------------------------------------------------------------------------
__global__ __launch_bounds__(256) void loss_kernel(const int* __restrict__ apB,
                                                   const int* __restrict__ anB,
                                                   float* __restrict__ out) {
    float acc = 0.0f;
    for (int i = threadIdx.x; i < N_ROWS; i += 256) {
        const float ap = __int_as_float(apB[i]);
        const float an = __int_as_float(anB[i]);
        acc += fmaxf(ap - an + MARGIN, 0.0f);
    }
    #pragma unroll
    for (int off = 16; off > 0; off >>= 1) acc += __shfl_xor(acc, off, 32);
    __shared__ float s[8];
    const int wave = threadIdx.x >> 5, lane = threadIdx.x & 31;
    if (lane == 0) s[wave] = acc;
    __syncthreads();
    if (threadIdx.x == 0) {
        float t = 0.0f;
        #pragma unroll
        for (int w = 0; w < 8; ++w) t += s[w];
        out[0] = t / (float)N_ROWS;
    }
}

// ---------------------------------------------------------------------------
extern "C" void kernel_launch(void* const* d_in, const int* in_sizes, int n_in,
                              void* d_out, int out_size, void* d_ws, size_t ws_size,
                              hipStream_t stream) {
    (void)in_sizes; (void)n_in; (void)out_size; (void)ws_size;
    const float* X = (const float*)d_in[0];
    const int*   Y = (const int*)d_in[1];
    float* out = (float*)d_out;

    char* ws = (char*)d_ws;
    _Float16* Xh = (_Float16*)ws;                       // 4096*2048*2 = 16 MiB
    const size_t xh_bytes = (size_t)N_ROWS * N_DIM * sizeof(_Float16);
    float* sq  = (float*)(ws + xh_bytes);               // 16 KiB
    int*   apB = (int*)(ws + xh_bytes + (size_t)N_ROWS * 4);
    int*   anB = (int*)(ws + xh_bytes + (size_t)N_ROWS * 8);

    prep_kernel<<<N_ROWS, 256, 0, stream>>>(X, Xh, sq, apB, anB);
    gram_kernel<<<dim3(N_ROWS / 128, N_ROWS / 128), 256, 0, stream>>>(Xh, sq, Y, apB, anB);
    loss_kernel<<<1, 256, 0, stream>>>(apB, anB, out);
}
